// EnhancedSAGEModel_54606214201943
// MI455X (gfx1250) — matrix operand order repaired and checked by reading, not compiled, not run on hardware
//
#include <hip/hip_runtime.h>
#include <hip/hip_bf16.h>

// ---------- types ----------
typedef __attribute__((ext_vector_type(16))) __bf16          v16bf;
typedef __attribute__((ext_vector_type(8)))  float           v8f;
typedef __attribute__((ext_vector_type(8)))  unsigned short  u16x8;
typedef __attribute__((ext_vector_type(16))) unsigned short  u16x16;

#define DEV __device__ __forceinline__

DEV unsigned short f2bf(float f) {
    return __builtin_bit_cast(unsigned short, (__bf16)f);
}

DEV v16bf mk16(u16x8 a, u16x8 b) {
    u16x16 t = __builtin_shufflevector(a, b, 0,1,2,3,4,5,6,7,8,9,10,11,12,13,14,15);
    return __builtin_bit_cast(v16bf, t);
}

// Build 16x32 bf16 A-fragment (lane holds K = k0+sub..+15 contiguous
// with sub = 16*(lane>=16)) directly from global fp32, with an
// optional per-row scale (folds the mean-aggregation division into the GEMM).
DEV v16bf load_afrag(const float* __restrict__ A, int row, int k0, int sub,
                     float scale, bool valid) {
    float buf[16];
    if (valid) {
        const float* p = A + (size_t)row * 128 + k0 + sub;
        float4 a = *(const float4*)(p);
        float4 b = *(const float4*)(p + 4);
        float4 c = *(const float4*)(p + 8);
        float4 d = *(const float4*)(p + 12);
        buf[0]=a.x;  buf[1]=a.y;  buf[2]=a.z;  buf[3]=a.w;
        buf[4]=b.x;  buf[5]=b.y;  buf[6]=b.z;  buf[7]=b.w;
        buf[8]=c.x;  buf[9]=c.y;  buf[10]=c.z; buf[11]=c.w;
        buf[12]=d.x; buf[13]=d.y; buf[14]=d.z; buf[15]=d.w;
    } else {
#pragma unroll
        for (int i = 0; i < 16; ++i) buf[i] = 0.0f;
    }
    v16bf r;
#pragma unroll
    for (int i = 0; i < 16; ++i) r[i] = (__bf16)(buf[i] * scale);
    return r;
}

// B-fragment (32x16 bf16, B[k][n] = W[n][k]) from LDS-staged bf16 weights:
// lane holds column n = ct*16 + (lane&15), K = k0 + 16*(lane>=16) .. +15
// -> 16 contiguous halfs -> two ds_load_b128.
DEV v16bf load_bfrag(const unsigned short* lds, int n, int k0, int ksub) {
    const u16x8* p = (const u16x8*)(lds + (size_t)n * 128 + k0 + ksub);
    return mk16(p[0], p[1]);
}

// ---------- GEMM: Out[r, :HOUT] = Ascaled@W^T (+ A2@W2^T) + bias (opt ReLU) ----
// A is [nrows x 128] fp32, W/W2 are [HOUT x 128] fp32. For DUAL, A is scaled
// per-row by 1/max(deg,1) (mean aggregation) and A2 is the raw features.
template <int HOUT, bool DUAL, bool RELU>
__global__ __launch_bounds__(128) void gemm_kernel(
    const float* __restrict__ A,  const float* __restrict__ W,
    const float* __restrict__ A2, const float* __restrict__ W2,
    const float* __restrict__ deg, const float* __restrict__ bias,
    float* __restrict__ Out, int nrows)
{
    constexpr int K = 128;
    constexpr int NCT = HOUT / 16;
    __shared__ unsigned short Wl[HOUT * K];
    __shared__ unsigned short W2l[DUAL ? HOUT * K : 1];

    const int tid = threadIdx.x;
    for (int i = tid; i < HOUT * K; i += 128) {
        Wl[i] = f2bf(W[i]);
        if (DUAL) W2l[i] = f2bf(W2[i]);
    }
    __syncthreads();

    const int wave = tid >> 5;
    const int lane = tid & 31;
    const int lh   = lane >> 4;      // half-wave id
    const int ll   = lane & 15;
    const int m0   = blockIdx.x * 64 + wave * 16;
    const int arow = m0 + ll;
    const bool valid = arow < nrows;

    float rdeg = 1.0f;
    if (DUAL) {
        float dg = valid ? deg[arow] : 1.0f;
        rdeg = 1.0f / fmaxf(dg, 1.0f);
    }
    const int sub  = lh * 16;
    const int ksub = lh * 16;

    v8f acc[NCT];
#pragma unroll
    for (int ct = 0; ct < NCT; ++ct)
#pragma unroll
        for (int r = 0; r < 8; ++r) acc[ct][r] = 0.0f;

#pragma unroll
    for (int kk = 0; kk < 4; ++kk) {
        const int k0 = kk * 32;
        v16bf af = load_afrag(A, arow, k0, sub, DUAL ? rdeg : 1.0f, valid);
#pragma unroll
        for (int ct = 0; ct < NCT; ++ct) {
            v16bf bf = load_bfrag(Wl, ct * 16 + ll, k0, ksub);
            acc[ct] = __builtin_amdgcn_wmma_f32_16x16x32_bf16(
                false, af, false, bf, (short)0, acc[ct], false, false);
        }
        if (DUAL) {
            v16bf af2 = load_afrag(A2, arow, k0, sub, 1.0f, valid);
#pragma unroll
            for (int ct = 0; ct < NCT; ++ct) {
                v16bf bf = load_bfrag(W2l, ct * 16 + ll, k0, ksub);
                acc[ct] = __builtin_amdgcn_wmma_f32_16x16x32_bf16(
                    false, af2, false, bf, (short)0, acc[ct], false, false);
            }
        }
    }

    // C/D layout: VGPR r -> row m0 + r + 8*lh, col = ct*16 + ll
#pragma unroll
    for (int ct = 0; ct < NCT; ++ct) {
        const int col = ct * 16 + ll;
        const float bv = bias[col];
#pragma unroll
        for (int r = 0; r < 8; ++r) {
            const int orow = m0 + r + lh * 8;
            if (orow < nrows) {
                float v = acc[ct][r] + bv;
                if (RELU) v = fmaxf(v, 0.0f);
                Out[(size_t)orow * HOUT + col] = v;
            }
        }
    }
}

// ---------- scatter: agg[dst] += X[src], deg[dst] += 1 (one wave per edge) ----
__global__ __launch_bounds__(256) void scatter_kernel(
    const float* __restrict__ X, const int* __restrict__ src,
    const int* __restrict__ dst, float* __restrict__ agg,
    float* __restrict__ deg, int E, int withdeg)
{
    const int e = blockIdx.x * 8 + (threadIdx.x >> 5);
    if (e >= E) return;
    const int lane = threadIdx.x & 31;
    const int s = src[e];
    const int d = dst[e];
    const float4 v = *(const float4*)(X + (size_t)s * 128 + lane * 4);
    float* p = agg + (size_t)d * 128 + lane * 4;
    __hip_atomic_fetch_add(p + 0, v.x, __ATOMIC_RELAXED, __HIP_MEMORY_SCOPE_AGENT);
    __hip_atomic_fetch_add(p + 1, v.y, __ATOMIC_RELAXED, __HIP_MEMORY_SCOPE_AGENT);
    __hip_atomic_fetch_add(p + 2, v.z, __ATOMIC_RELAXED, __HIP_MEMORY_SCOPE_AGENT);
    __hip_atomic_fetch_add(p + 3, v.w, __ATOMIC_RELAXED, __HIP_MEMORY_SCOPE_AGENT);
    if (withdeg && lane == 0)
        __hip_atomic_fetch_add(deg + d, 1.0f, __ATOMIC_RELAXED, __HIP_MEMORY_SCOPE_AGENT);
}

// ---------- column sum / sum-of-squares over [n x 128] ----------
__global__ __launch_bounds__(128) void colstats_kernel(
    const float* __restrict__ S, float* __restrict__ stats, int n)
{
    const int c = threadIdx.x;  // 0..127
    float sum = 0.0f, sq = 0.0f;
    for (int r = blockIdx.x; r < n; r += gridDim.x) {
        float v = S[(size_t)r * 128 + c];
        sum += v; sq += v * v;
    }
    __hip_atomic_fetch_add(stats + c,       sum, __ATOMIC_RELAXED, __HIP_MEMORY_SCOPE_AGENT);
    __hip_atomic_fetch_add(stats + 128 + c, sq,  __ATOMIC_RELAXED, __HIP_MEMORY_SCOPE_AGENT);
}

// ---------- fused BN(train stats) -> ReLU -> + residual ----------
__global__ __launch_bounds__(256) void bn_relu_add_kernel(
    const float* __restrict__ S, const float* __restrict__ stats,
    const float* __restrict__ g, const float* __restrict__ b,
    const float* __restrict__ res, float* __restrict__ out, int n)
{
    const size_t idx = (size_t)blockIdx.x * 256 + threadIdx.x;
    if (idx >= (size_t)n * 128) return;
    const int c = (int)(idx & 127);
    const float invn = 1.0f / (float)n;
    const float m = stats[c] * invn;
    const float var = stats[128 + c] * invn - m * m;
    float v = (S[idx] - m) * rsqrtf(var + 1e-5f) * g[c] + b[c];
    v = fmaxf(v, 0.0f);
    out[idx] = v + res[idx];
}

__global__ __launch_bounds__(256) void zero_kernel(float* __restrict__ p, size_t count)
{
    const size_t i = (size_t)blockIdx.x * 256 + threadIdx.x;
    if (i < count) p[i] = 0.0f;
}

// ---------- host ----------
static inline unsigned int cdiv(size_t a, size_t b) { return (unsigned int)((a + b - 1) / b); }

extern "C" void kernel_launch(void* const* d_in, const int* in_sizes, int n_in,
                              void* d_out, int out_size, void* d_ws, size_t ws_size,
                              hipStream_t stream) {
    const float* x     = (const float*)d_in[0];
    const int*   ei    = (const int*)d_in[1];
    const float* s1_wl = (const float*)d_in[2];
    const float* s1_wr = (const float*)d_in[3];
    const float* s1_b  = (const float*)d_in[4];
    const float* s2_wl = (const float*)d_in[5];
    const float* s2_wr = (const float*)d_in[6];
    const float* s2_b  = (const float*)d_in[7];
    const float* s3_wl = (const float*)d_in[8];
    const float* s3_wr = (const float*)d_in[9];
    const float* s3_b  = (const float*)d_in[10];
    const float* bn1_g = (const float*)d_in[11];
    const float* bn1_b = (const float*)d_in[12];
    const float* bn2_g = (const float*)d_in[13];
    const float* bn2_b = (const float*)d_in[14];
    const float* res1_w = (const float*)d_in[15];
    const float* res1_b = (const float*)d_in[16];
    const float* res2_w = (const float*)d_in[17];
    const float* res2_b = (const float*)d_in[18];
    const float* ff1_w  = (const float*)d_in[19];
    const float* ff1_b  = (const float*)d_in[20];
    const float* ff2_w  = (const float*)d_in[21];
    const float* ff2_b  = (const float*)d_in[22];
    const float* clf_w  = (const float*)d_in[23];
    const float* clf_b  = (const float*)d_in[24];

    const int N = in_sizes[0] / 128;
    const int E = in_sizes[1] / 2;
    const int* src = ei;
    const int* dst = ei + E;

    const size_t NH   = (size_t)N * 128;
    const size_t PADN = ((size_t)N + 2047) & ~(size_t)2047;
    float* ws    = (float*)d_ws;
    float* deg   = ws;                  // [PADN]
    float* agg   = deg + PADN;          // [NH]
    float* sbuf  = agg + NH;            // [NH]
    float* resb  = sbuf + NH;           // [NH]
    float* hbuf  = resb + NH;           // [NH]
    float* stats = hbuf + NH;           // [256]

    const dim3 blk256(256), blk128(128);
    const unsigned int gScat = cdiv((size_t)E, 8);
    const unsigned int gGemm = cdiv((size_t)N, 64);
    const unsigned int gElem = cdiv(NH, 256);

    // ---- block 1 ----
    zero_kernel<<<cdiv(PADN + NH, 256), blk256, 0, stream>>>(deg, PADN + NH);
    scatter_kernel<<<gScat, blk256, 0, stream>>>(x, src, dst, agg, deg, E, 1);
    gemm_kernel<128, false, false><<<gGemm, blk128, 0, stream>>>(
        x, res1_w, nullptr, nullptr, nullptr, res1_b, resb, N);
    gemm_kernel<128, true, false><<<gGemm, blk128, 0, stream>>>(
        agg, s1_wl, x, s1_wr, deg, s1_b, sbuf, N);
    zero_kernel<<<1, blk256, 0, stream>>>(stats, 256);
    colstats_kernel<<<1024, blk128, 0, stream>>>(sbuf, stats, N);
    bn_relu_add_kernel<<<gElem, blk256, 0, stream>>>(sbuf, stats, bn1_g, bn1_b, resb, hbuf, N);

    // ---- block 2 ----
    zero_kernel<<<cdiv(NH, 256), blk256, 0, stream>>>(agg, NH);
    scatter_kernel<<<gScat, blk256, 0, stream>>>(hbuf, src, dst, agg, nullptr, E, 0);
    gemm_kernel<128, false, false><<<gGemm, blk128, 0, stream>>>(
        hbuf, res2_w, nullptr, nullptr, nullptr, res2_b, resb, N);
    gemm_kernel<128, true, false><<<gGemm, blk128, 0, stream>>>(
        agg, s2_wl, hbuf, s2_wr, deg, s2_b, sbuf, N);
    zero_kernel<<<1, blk256, 0, stream>>>(stats, 256);
    colstats_kernel<<<1024, blk128, 0, stream>>>(sbuf, stats, N);
    bn_relu_add_kernel<<<gElem, blk256, 0, stream>>>(sbuf, stats, bn2_g, bn2_b, resb, hbuf, N);

    // ---- block 3 + FFN head ----
    zero_kernel<<<cdiv(NH, 256), blk256, 0, stream>>>(agg, NH);
    scatter_kernel<<<gScat, blk256, 0, stream>>>(hbuf, src, dst, agg, nullptr, E, 0);
    gemm_kernel<128, true, false><<<gGemm, blk128, 0, stream>>>(
        agg, s3_wl, hbuf, s3_wr, deg, s3_b, sbuf, N);
    gemm_kernel<128, false, true><<<gGemm, blk128, 0, stream>>>(
        sbuf, ff1_w, nullptr, nullptr, nullptr, ff1_b, resb, N);     // relu(ffn1)
    gemm_kernel<128, false, false><<<gGemm, blk128, 0, stream>>>(
        resb, ff2_w, nullptr, nullptr, nullptr, ff2_b, sbuf, N);     // ffn2
    gemm_kernel<64, false, false><<<gGemm, blk128, 0, stream>>>(
        sbuf, clf_w, nullptr, nullptr, nullptr, clf_b, (float*)d_out, N);
    (void)n_in; (void)out_size; (void)ws_size;
}
